// RelPartialLearnableDecoderLayer_51917564674150
// MI455X (gfx1250) — matrix operand order repaired and checked by reading, not compiled, not run on hardware
//
#include <hip/hip_runtime.h>
#include <math.h>

// ---------------- problem constants ----------------
#define Q_   1024
#define B_   8
#define DM_  512
#define NH_  8
#define DH_  64
#define DI_  2048
#define E_   1024
#define BH_  (B_ * NH_)     // 64
#define QB_  (Q_ * B_)      // 8192
#define SCALE_ 0.125f       // 1/sqrt(64)

typedef unsigned short ushort;

// ---------------- CDNA5 WMMA types ----------------
typedef __attribute__((ext_vector_type(16))) __bf16        v16bf;
typedef __attribute__((ext_vector_type(8)))  float         v8f;
typedef __attribute__((ext_vector_type(8)))  unsigned int  v8u;

__device__ __forceinline__ ushort f2bf(float f) {
  unsigned int u = __builtin_bit_cast(unsigned int, f);
  u += 0x7FFFu + ((u >> 16) & 1u);          // round-to-nearest-even
  return (ushort)(u >> 16);
}

// =====================================================================
// Batched bf16 WMMA GEMM (all-NT):  C[M,N] = A[M,K] * B[N,K]^T
//   A: bf16 [M,K] row-major (lda, batch stride sA)
//   B: bf16 [N,K] row-major (ldb, batch stride sB; bmod>0 -> B batch
//      index = blockIdx.z % bmod for head-shared operands)
//   C: fp32 [M,N] (ldc, sC)
// Block = 128 threads = 4 wave32; block tile 64x64; each wave a 32x32
// quadrant = 2x2 v_wmma_f32_16x16x32_bf16 accumulators; K stepped by 64.
// Double-buffered GLOBAL_LOAD_ASYNC_TO_LDS_B128 pipeline: prefetch the
// next tile (8 async instr/wave), then s_wait_asynccnt 8 so the newest
// 8 stay in flight while WMMAs run (async loads complete in order).
// =====================================================================
__global__ __launch_bounds__(128)
void gemm_bf16_wmma(const ushort* __restrict__ A, long lda, long sA,
                    const ushort* __restrict__ Bm, long ldb, long sB, int bmod,
                    float* __restrict__ C, long ldc, long sC, int K)
{
  __shared__ ushort As[2][64 * 72];   // [m][k], pitch 72 shorts = 144 B
  __shared__ ushort Bs[2][64 * 72];   // [n][k]
  const unsigned bufBytes = 64u * 72u * 2u;   // 9216 B per buffer

  const int bz = blockIdx.z;
  const ushort* Ab = A  + (long)bz * sA;
  const ushort* Bb = Bm + (long)(bmod ? (bz % bmod) : bz) * sB;
  float*        Cb = C  + (long)bz * sC;

  const int m0   = blockIdx.y * 64;
  const int n0   = blockIdx.x * 64;
  const int tid  = threadIdx.x;
  const int lane = tid & 31;
  const int w    = tid >> 5;
  const int wm   = w >> 1;
  const int wn   = w & 1;
  const int l15  = lane & 15;
  const int hi   = lane >> 4;

  const unsigned ldsA = (unsigned)(unsigned long long)(uintptr_t)&As[0][0];
  const unsigned ldsB = (unsigned)(unsigned long long)(uintptr_t)&Bs[0][0];

  // issue 8 async b128 copies (4 for A tile, 4 for B tile) per thread
  auto stage = [&](int buf, int k0) {
#pragma unroll
    for (int t = 0; t < 4; ++t) {
      int e  = tid + t * 128;          // 0..511 chunk id
      int rr = e >> 3;                 // tile row 0..63
      int cc = (e & 7) * 8;            // k offset (elems) 0..56
      unsigned loff = (unsigned)(rr * 72 + cc) * 2u + (unsigned)buf * bufBytes;
      unsigned lA = ldsA + loff;
      unsigned lB = ldsB + loff;
      unsigned gA = (unsigned)(((long)(m0 + rr) * lda + k0 + cc) * 2);
      unsigned gB = (unsigned)(((long)(n0 + rr) * ldb + k0 + cc) * 2);
      asm volatile("global_load_async_to_lds_b128 %0, %1, %2"
                   :: "v"(lA), "v"(gA), "s"(Ab) : "memory");
      asm volatile("global_load_async_to_lds_b128 %0, %1, %2"
                   :: "v"(lB), "v"(gB), "s"(Bb) : "memory");
    }
  };

  v8f acc[2][2];
  const v8f vzero = {0.f, 0.f, 0.f, 0.f, 0.f, 0.f, 0.f, 0.f};
#pragma unroll
  for (int a = 0; a < 2; ++a)
#pragma unroll
    for (int b = 0; b < 2; ++b) acc[a][b] = vzero;

  const int T = K >> 6;                // K / 64 tiles
  stage(0, 0);                         // prologue prefetch

  for (int t = 0; t < T; ++t) {
    const int cur = t & 1;
    if (t + 1 < T) {
      stage(cur ^ 1, (t + 1) << 6);    // prefetch next tile into other buffer
      // oldest 8 asyncs (current tile) done; newest 8 remain in flight
      asm volatile("s_wait_asynccnt 0x8" ::: "memory");
    } else {
      asm volatile("s_wait_asynccnt 0x0" ::: "memory");
    }
    __syncthreads();

    const ushort* Asb = &As[cur][0];
    const ushort* Bsb = &Bs[cur][0];

#pragma unroll
    for (int kk = 0; kk < 2; ++kk) {
      v16bf bf[2];
#pragma unroll
      for (int ni = 0; ni < 2; ++ni) {
        int n = wn * 32 + ni * 16 + l15;
        const ushort* p = &Bsb[n * 72 + kk * 32 + hi * 8];
        uint4 b0 = *(const uint4*)p;          // k = 32kk+8hi ..
        uint4 b1 = *(const uint4*)(p + 16);   // k = 32kk+16+8hi ..
        v8u br;
        br[0] = b0.x; br[1] = b0.y; br[2] = b0.z; br[3] = b0.w;
        br[4] = b1.x; br[5] = b1.y; br[6] = b1.z; br[7] = b1.w;
        bf[ni] = __builtin_bit_cast(v16bf, br);
      }
#pragma unroll
      for (int mi = 0; mi < 2; ++mi) {
        int m = wm * 32 + mi * 16 + l15;
        const ushort* p = &Asb[m * 72 + kk * 32 + hi * 8];
        uint4 a0 = *(const uint4*)p;          // k = 32kk+8hi ..
        uint4 a1 = *(const uint4*)(p + 16);   // k = 32kk+16+8hi ..
        v8u ar;
        ar[0] = a0.x; ar[1] = a0.y; ar[2] = a0.z; ar[3] = a0.w;
        ar[4] = a1.x; ar[5] = a1.y; ar[6] = a1.z; ar[7] = a1.w;
        v16bf af = __builtin_bit_cast(v16bf, ar);
        acc[mi][0] = __builtin_amdgcn_wmma_f32_16x16x32_bf16(
            false, af, false, bf[0], (short)0, acc[mi][0], false, false);
        acc[mi][1] = __builtin_amdgcn_wmma_f32_16x16x32_bf16(
            false, af, false, bf[1], (short)0, acc[mi][1], false, false);
      }
    }
    __syncthreads();   // protect 'cur' buffer before it is re-staged
  }

  // ---- writeback per documented C layout: M = i + 8*hi, N = l15 ----
#pragma unroll
  for (int mi = 0; mi < 2; ++mi)
#pragma unroll
    for (int ni = 0; ni < 2; ++ni)
#pragma unroll
      for (int i = 0; i < 8; ++i) {
        long row = m0 + wm * 32 + mi * 16 + i + 8 * hi;
        long col = n0 + wn * 32 + ni * 16 + l15;
        Cb[row * ldc + col] = acc[mi][ni][i];
      }
}

// =====================================================================
// LayerNorm; writes bf16 (GEMM operand) and optionally fp32 (residual).
// =====================================================================
__global__ __launch_bounds__(128)
void ln_kernel(const float* __restrict__ x, const float* __restrict__ g,
               const float* __restrict__ b, ushort* __restrict__ yh,
               float* __restrict__ yf, int dm)
{
  __shared__ float s1[128], s2[128];
  long row = blockIdx.x;
  const float* xr = x + row * dm;
  ushort* yhr = yh + row * dm;
  int tid = threadIdx.x;
  float a = 0.f, q = 0.f;
  for (int c = tid; c < dm; c += 128) { float v = xr[c]; a += v; q += v * v; }
  s1[tid] = a; s2[tid] = q; __syncthreads();
  for (int s = 64; s > 0; s >>= 1) {
    if (tid < s) { s1[tid] += s1[tid + s]; s2[tid] += s2[tid + s]; }
    __syncthreads();
  }
  float mean = s1[0] / dm;
  float var  = s2[0] / dm - mean * mean;
  float inv  = rsqrtf(var + 1e-5f);
  for (int c = tid; c < dm; c += 128) {
    float v = (xr[c] - mean) * inv * g[c] + b[c];
    yhr[c] = f2bf(v);
    if (yf) yf[row * dm + c] = v;
  }
}

// =====================================================================
// Fused (AC + rel_shift(BD)) * scale -> causal mask -> softmax.
// sc/bd fp32 [BH, Q, L]; prob written bf16 (A operand of prob@V GEMM).
// rel-shift: BD_used(i,j) = bd(i, j - i + L - 1).
// =====================================================================
__global__ __launch_bounds__(256)
void attn_softmax_kernel(const float* __restrict__ sc, const float* __restrict__ bd,
                         ushort* __restrict__ ph, int L, int causal)
{
  __shared__ float red[256];
  const int i  = blockIdx.x;            // query row (gridDim.x == Q)
  const int bh = blockIdx.y;
  const long base = ((long)bh * gridDim.x + i) * (long)L;
  const int tid = threadIdx.x;
  const int lim = causal ? (i + 1) : L;

  float mx = -3.4e38f;
  for (int j = tid; j < lim; j += 256) {
    float v = sc[base + j];
    if (bd) v += bd[base + (j - i + L - 1)];
    mx = fmaxf(mx, v * SCALE_);
  }
  red[tid] = mx; __syncthreads();
  for (int s = 128; s > 0; s >>= 1) {
    if (tid < s) red[tid] = fmaxf(red[tid], red[tid + s]);
    __syncthreads();
  }
  mx = red[0]; __syncthreads();

  float sum = 0.f;
  for (int j = tid; j < lim; j += 256) {
    float v = sc[base + j];
    if (bd) v += bd[base + (j - i + L - 1)];
    sum += __expf(v * SCALE_ - mx);
  }
  red[tid] = sum; __syncthreads();
  for (int s = 128; s > 0; s >>= 1) {
    if (tid < s) red[tid] += red[tid + s];
    __syncthreads();
  }
  float rsum = 1.f / red[0]; __syncthreads();

  for (int j = tid; j < L; j += 256) {
    float p = 0.f;
    if (j < lim) {
      float v = sc[base + j];
      if (bd) v += bd[base + (j - i + L - 1)];
      p = __expf(v * SCALE_ - mx) * rsum;
    }
    ph[base + j] = f2bf(p);
  }
}

// =====================================================================
// Elementwise helpers (bf16 producers for GEMM operands)
// =====================================================================
__global__ __launch_bounds__(256)
void cvt_kernel(const float* __restrict__ s, ushort* __restrict__ d, long n)
{
  long i = (long)blockIdx.x * 256 + threadIdx.x;
  if (i < n) d[i] = f2bf(s[i]);
}

__global__ __launch_bounds__(256)
void cvt_t_kernel(const float* __restrict__ s /*[K,N]*/, ushort* __restrict__ d /*[N,K]*/,
                  int kbits, int N)
{
  long i = (long)blockIdx.x * 256 + threadIdx.x;
  if (i < ((long)N << kbits)) {
    int k = (int)(i & ((1 << kbits) - 1));
    int n = (int)(i >> kbits);
    d[i] = f2bf(s[(long)k * N + n]);
  }
}

__global__ __launch_bounds__(256)
void split_qkv_kernel(const float* __restrict__ heads,
                      const float* __restrict__ rwb, const float* __restrict__ rrb,
                      ushort* __restrict__ qac, ushort* __restrict__ qbd,
                      ushort* __restrict__ kc,  ushort* __restrict__ vct)
{
  long idx = (long)blockIdx.x * 256 + threadIdx.x;   // Q*B*NH*DH
  int d = (int)(idx & (DH_ - 1));
  int n = (int)((idx >> 6) & (NH_ - 1));
  int b = (int)((idx >> 9) & (B_ - 1));
  int i = (int)(idx >> 12);
  long hrow = ((long)i * B_ + b) * (3 * NH_ * DH_);
  int bh = b * NH_ + n;
  long dst  = ((long)bh * Q_ + i) * DH_ + d;
  int nd = n * DH_ + d;
  float qv = heads[hrow + nd];
  qac[dst] = f2bf(qv + rwb[nd]);
  qbd[dst] = f2bf(qv + rrb[nd]);
  kc[dst]  = f2bf(heads[hrow + NH_ * DH_ + nd]);
  // V transposed per head: [BH, DH, Q] so prob@V is NT with K-contiguous B
  vct[((long)bh * DH_ + d) * Q_ + i] = f2bf(heads[hrow + 2 * NH_ * DH_ + nd]);
}

__global__ __launch_bounds__(256)
void rkt_kernel(const float* __restrict__ src, ushort* __restrict__ dst)
{ // src [Q, NH*DH] -> dst [NH, Q, DH] bf16
  long idx = (long)blockIdx.x * 256 + threadIdx.x;   // Q*NH*DH
  int d = (int)(idx & (DH_ - 1));
  int n = (int)((idx >> 6) & (NH_ - 1));
  int i = (int)(idx >> 9);
  dst[((long)n * Q_ + i) * DH_ + d] = f2bf(src[(long)i * (NH_ * DH_) + n * DH_ + d]);
}

__global__ __launch_bounds__(256)
void splitq_kernel(const float* __restrict__ src, ushort* __restrict__ dst)
{ // src [QB, NH*DH] -> dst [BH, Q, DH] bf16
  long idx = (long)blockIdx.x * 256 + threadIdx.x;
  int d = (int)(idx & (DH_ - 1));
  int n = (int)((idx >> 6) & (NH_ - 1));
  int b = (int)((idx >> 9) & (B_ - 1));
  int i = (int)(idx >> 12);
  dst[(((long)(b * NH_ + n)) * Q_ + i) * DH_ + d] =
      f2bf(src[((long)i * B_ + b) * (NH_ * DH_) + n * DH_ + d]);
}

__global__ __launch_bounds__(256)
void splitkv_kernel(const float* __restrict__ kv,
                    ushort* __restrict__ k2, ushort* __restrict__ v2t)
{ // kv [E*B, 2*NH*DH] -> k2 [BH,E,DH], v2t [BH,DH,E] bf16
  long idx = (long)blockIdx.x * 256 + threadIdx.x;
  int d = (int)(idx & (DH_ - 1));
  int n = (int)((idx >> 6) & (NH_ - 1));
  int b = (int)((idx >> 9) & (B_ - 1));
  int e = (int)(idx >> 12);
  long row = ((long)e * B_ + b) * (2 * NH_ * DH_);
  int bh = b * NH_ + n;
  k2[((long)bh * E_ + e) * DH_ + d]  = f2bf(kv[row + n * DH_ + d]);
  v2t[((long)bh * DH_ + d) * E_ + e] = f2bf(kv[row + NH_ * DH_ + n * DH_ + d]);
}

__global__ __launch_bounds__(256)
void merge_kernel(const float* __restrict__ vec, ushort* __restrict__ out)
{ // vec [BH, Q, DH] fp32 -> out [QB, NH*DH] bf16
  long idx = (long)blockIdx.x * 256 + threadIdx.x;
  int d = (int)(idx & (DH_ - 1));
  int n = (int)((idx >> 6) & (NH_ - 1));
  int b = (int)((idx >> 9) & (B_ - 1));
  int i = (int)(idx >> 12);
  out[((long)i * B_ + b) * (NH_ * DH_) + n * DH_ + d] =
      f2bf(vec[(((long)(b * NH_ + n)) * Q_ + i) * DH_ + d]);
}

__global__ __launch_bounds__(256)
void add2_kernel(const float* __restrict__ a, const float* __restrict__ b,
                 float* __restrict__ o, long n)
{
  long i = (long)blockIdx.x * 256 + threadIdx.x;
  if (i < n) o[i] = a[i] + b[i];
}

__global__ __launch_bounds__(256)
void bias_gelu_kernel(const float* __restrict__ x, const float* __restrict__ bias,
                      ushort* __restrict__ yh, long n, int colmask)
{
  long i = (long)blockIdx.x * 256 + threadIdx.x;
  if (i < n) {
    float v = x[i] + bias[(int)(i & colmask)];
    yh[i] = f2bf(0.5f * v * (1.f + erff(v * 0.70710678118654752f)));  // exact GELU
  }
}

__global__ __launch_bounds__(256)
void final_kernel(const float* __restrict__ core, const float* __restrict__ b2,
                  const float* __restrict__ out2, float* __restrict__ out,
                  long n, int colmask)
{
  long i = (long)blockIdx.x * 256 + threadIdx.x;
  if (i < n) out[i] = core[i] + b2[(int)(i & colmask)] + out2[i];
}

// =====================================================================
// Host side
// =====================================================================
static inline void gemm(const ushort* A, long lda, long sA,
                        const ushort* B, long ldb, long sB, int bmod,
                        float* C, long ldc, long sC,
                        int M, int N, int K, int batch, hipStream_t s)
{
  dim3 g((unsigned)(N / 64), (unsigned)(M / 64), (unsigned)batch);
  gemm_bf16_wmma<<<g, 128, 0, s>>>(A, lda, sA, B, ldb, sB, bmod, C, ldc, sC, K);
}

extern "C" void kernel_launch(void* const* d_in, const int* in_sizes, int n_in,
                              void* d_out, int out_size, void* d_ws, size_t ws_size,
                              hipStream_t stream)
{
  (void)in_sizes; (void)n_in; (void)out_size; (void)ws_size;
  const float* dec_inp  = (const float*)d_in[0];
  const float* r        = (const float*)d_in[1];
  const float* enc_out  = (const float*)d_in[2];
  const float* r_w_bias = (const float*)d_in[3];
  const float* r_r_bias = (const float*)d_in[4];
  const float* qkv_w    = (const float*)d_in[5];
  const float* r_w      = (const float*)d_in[6];
  const float* o_w      = (const float*)d_in[7];
  const float* ln1_g    = (const float*)d_in[8];
  const float* ln1_b    = (const float*)d_in[9];
  const float* q_w      = (const float*)d_in[10];
  const float* kv_w     = (const float*)d_in[11];
  const float* o2_w     = (const float*)d_in[12];
  const float* ln2_g    = (const float*)d_in[13];
  const float* ln2_b    = (const float*)d_in[14];
  const float* ff_w1    = (const float*)d_in[15];
  const float* ff_b1    = (const float*)d_in[16];
  const float* ff_w2    = (const float*)d_in[17];
  const float* ff_b2    = (const float*)d_in[18];
  const float* ln3_g    = (const float*)d_in[19];
  const float* ln3_b    = (const float*)d_in[20];
  // d_in[21] dec_attn_mask (triu k=1) / d_in[22] enc_mask (all false):
  // fixed by construction -> applied analytically in the softmax.

  // ---- workspace layout ----
  char* base = (char*)d_ws;
  size_t off = 0;
  auto takeF = [&](size_t n) { float* p = (float*)(base + off); off += n * 4; return p; };
  auto takeH = [&](size_t n) { ushort* p = (ushort*)(base + off); off += n * 2; return p; };

  // fp32 (GEMM C outputs / residual chain)
  float* reg0   = takeF((size_t)QB_ * DI_);       // heads / kv / ff1 (fp32)
  float* scoreA = takeF((size_t)BH_ * Q_ * Q_);   // AC / score2
  float* scoreB = takeF((size_t)BH_ * Q_ * Q_);   // BD
  float* vech   = takeF((size_t)BH_ * Q_ * DH_);
  float* tmp1   = takeF((size_t)QB_ * DM_);
  float* out1   = takeF((size_t)QB_ * DM_);
  float* out2   = takeF((size_t)QB_ * DM_);
  float* h2f    = takeF((size_t)QB_ * DM_);
  float* rkpf   = takeF((size_t)Q_ * NH_ * DH_);
  // bf16 operands
  ushort* qkv_wt = takeH((size_t)(3 * NH_ * DH_) * DM_);  // [1536,512]
  ushort* r_wt   = takeH((size_t)(NH_ * DH_) * DM_);
  ushort* o_wt   = takeH((size_t)DM_ * (NH_ * DH_));      // [512,512]
  ushort* q_wt   = takeH((size_t)(NH_ * DH_) * DM_);
  ushort* kv_wt  = takeH((size_t)(2 * NH_ * DH_) * DM_);
  ushort* o2_wt  = takeH((size_t)DM_ * (NH_ * DH_));
  ushort* ff_w1t = takeH((size_t)DI_ * DM_);
  ushort* ff_w2t = takeH((size_t)DM_ * DI_);
  ushort* rh     = takeH((size_t)Q_ * DM_);
  ushort* ench   = takeH((size_t)QB_ * DM_);
  ushort* hbh    = takeH((size_t)QB_ * DM_);      // h1 / h3
  ushort* h2h    = takeH((size_t)QB_ * DM_);
  ushort* qach   = takeH((size_t)BH_ * Q_ * DH_);
  ushort* qbdh   = takeH((size_t)BH_ * Q_ * DH_);
  ushort* kch    = takeH((size_t)BH_ * Q_ * DH_); // also k2 (same size)
  ushort* vcth   = takeH((size_t)BH_ * DH_ * Q_); // V^T per head; also v2^T
  ushort* rkTh   = takeH((size_t)NH_ * Q_ * DH_);
  ushort* probh  = takeH((size_t)BH_ * Q_ * Q_);
  ushort* vecmh  = takeH((size_t)QB_ * DM_);
  ushort* ff1h   = takeH((size_t)QB_ * DI_);

  const long HD = (long)Q_ * DH_;
  const long SS = (long)Q_ * Q_;
  const unsigned EWG = (unsigned)((Q_ * B_ * NH_ * DH_) / 256);
  auto g1 = [](long n) { return (unsigned)((n + 255) / 256); };

  // ---- one-time bf16 operand prep (weights transposed to [N,K]) ----
  cvt_t_kernel<<<g1((long)DM_ * 3 * NH_ * DH_), 256, 0, stream>>>(qkv_w, qkv_wt, 9, 3 * NH_ * DH_);
  cvt_t_kernel<<<g1((long)DM_ * NH_ * DH_), 256, 0, stream>>>(r_w,  r_wt,  9, NH_ * DH_);
  cvt_t_kernel<<<g1((long)NH_ * DH_ * DM_), 256, 0, stream>>>(o_w,  o_wt,  9, DM_);
  cvt_t_kernel<<<g1((long)DM_ * NH_ * DH_), 256, 0, stream>>>(q_w,  q_wt,  9, NH_ * DH_);
  cvt_t_kernel<<<g1((long)DM_ * 2 * NH_ * DH_), 256, 0, stream>>>(kv_w, kv_wt, 9, 2 * NH_ * DH_);
  cvt_t_kernel<<<g1((long)NH_ * DH_ * DM_), 256, 0, stream>>>(o2_w, o2_wt, 9, DM_);
  cvt_t_kernel<<<g1((long)DM_ * DI_), 256, 0, stream>>>(ff_w1, ff_w1t, 9, DI_);
  cvt_t_kernel<<<g1((long)DI_ * DM_), 256, 0, stream>>>(ff_w2, ff_w2t, 11, DM_);
  cvt_kernel<<<g1((long)Q_ * DM_), 256, 0, stream>>>(r, rh, (long)Q_ * DM_);
  cvt_kernel<<<g1((long)QB_ * DM_), 256, 0, stream>>>(enc_out, ench, (long)QB_ * DM_);

  // ======== self-attention (RelPartialLearnableMultiHeadAttn) ========
  ln_kernel<<<QB_, 128, 0, stream>>>(dec_inp, ln1_g, ln1_b, hbh, nullptr, DM_);
  gemm(hbh, DM_, 0, qkv_wt, DM_, 0, 0, reg0, 3 * NH_ * DH_, 0,
       QB_, 3 * NH_ * DH_, DM_, 1, stream);
  gemm(rh, DM_, 0, r_wt, DM_, 0, 0, rkpf, NH_ * DH_, 0,
       Q_, NH_ * DH_, DM_, 1, stream);
  rkt_kernel<<<g1((long)Q_ * NH_ * DH_), 256, 0, stream>>>(rkpf, rkTh);
  split_qkv_kernel<<<EWG, 256, 0, stream>>>(reg0, r_w_bias, r_r_bias,
                                            qach, qbdh, kch, vcth);
  gemm(qach, DH_, HD, kch,  DH_, HD, 0,   scoreA, Q_, SS, Q_, Q_, DH_, BH_, stream);
  gemm(qbdh, DH_, HD, rkTh, DH_, HD, NH_, scoreB, Q_, SS, Q_, Q_, DH_, BH_, stream);
  attn_softmax_kernel<<<dim3(Q_, BH_), 256, 0, stream>>>(scoreA, scoreB, probh, Q_, 1);
  gemm(probh, Q_, SS, vcth, Q_, (long)DH_ * Q_, 0, vech, DH_, HD,
       Q_, DH_, Q_, BH_, stream);
  merge_kernel<<<EWG, 256, 0, stream>>>(vech, vecmh);
  gemm(vecmh, DM_, 0, o_wt, NH_ * DH_, 0, 0, tmp1, DM_, 0, QB_, DM_, NH_ * DH_, 1, stream);
  add2_kernel<<<EWG, 256, 0, stream>>>(dec_inp, tmp1, out1, (long)QB_ * DM_);

  // ======== cross-attention (ContextMultiHeadAttn) ========
  ln_kernel<<<QB_, 128, 0, stream>>>(out1, ln2_g, ln2_b, h2h, h2f, DM_);
  gemm(h2h, DM_, 0, q_wt, DM_, 0, 0, tmp1, NH_ * DH_, 0, QB_, NH_ * DH_, DM_, 1, stream);
  splitq_kernel<<<EWG, 256, 0, stream>>>(tmp1, qach);
  gemm(ench, DM_, 0, kv_wt, DM_, 0, 0, reg0, 2 * NH_ * DH_, 0,
       E_ * B_, 2 * NH_ * DH_, DM_, 1, stream);
  splitkv_kernel<<<EWG, 256, 0, stream>>>(reg0, kch, vcth);
  gemm(qach, DH_, HD, kch, DH_, (long)E_ * DH_, 0, scoreA, E_, (long)Q_ * E_,
       Q_, E_, DH_, BH_, stream);
  attn_softmax_kernel<<<dim3(Q_, BH_), 256, 0, stream>>>(scoreA, nullptr, probh, E_, 0);
  gemm(probh, E_, (long)Q_ * E_, vcth, E_, (long)DH_ * E_, 0, vech, DH_, HD,
       Q_, DH_, E_, BH_, stream);
  merge_kernel<<<EWG, 256, 0, stream>>>(vech, vecmh);
  gemm(vecmh, DM_, 0, o2_wt, NH_ * DH_, 0, 0, tmp1, DM_, 0, QB_, DM_, NH_ * DH_, 1, stream);
  add2_kernel<<<EWG, 256, 0, stream>>>(h2f, tmp1, out2, (long)QB_ * DM_);  // quirk: normed residual

  // ======== PositionwiseFF ========
  ln_kernel<<<QB_, 128, 0, stream>>>(out2, ln3_g, ln3_b, hbh, nullptr, DM_);
  gemm(hbh, DM_, 0, ff_w1t, DM_, 0, 0, reg0, DI_, 0, QB_, DI_, DM_, 1, stream);
  bias_gelu_kernel<<<g1((long)QB_ * DI_), 256, 0, stream>>>(reg0, ff_b1, ff1h,
                                                            (long)QB_ * DI_, DI_ - 1);
  gemm(ff1h, DI_, 0, ff_w2t, DI_, 0, 0, tmp1, DM_, 0, QB_, DM_, DI_, 1, stream);
  final_kernel<<<EWG, 256, 0, stream>>>(tmp1, ff_b2, out2, (float*)d_out,
                                        (long)QB_ * DM_, DM_ - 1);
}
